// DSRMM_919123001376
// MI455X (gfx1250) — matrix operand order repaired and checked by reading, not compiled, not run on hardware
//
#include <hip/hip_runtime.h>

typedef __attribute__((ext_vector_type(16))) _Float16 v16h;
typedef __attribute__((ext_vector_type(8)))  _Float16 v8h;
typedef __attribute__((ext_vector_type(8)))  float    v8f;

#define B_   64
#define Q_   10
#define L_   200
#define C_   50
#define CL_  8
#define E_   300
#define EP_  320   // E padded to 320 (10 chunks of 32)
#define D_   250
#define DP_  256   // D padded
#define ROWS_ 272  // 16 query rows + 256 doc rows
#define NP_  304   // uni output channels padded (19 N-tiles)

#define CONCAT16(lo, hi) __builtin_shufflevector((lo), (hi), 0,1,2,3,4,5,6,7,8,9,10,11,12,13,14,15)

// ---------------------------------------------------------------------------
// 1. Embedding gather: eq16 [B][16][320], ed16 [B][256][320], fp16, zero-padded
// ---------------------------------------------------------------------------
__global__ __launch_bounds__(320)
void prep_embed(const int* __restrict__ bq, const int* __restrict__ bd,
                const int* __restrict__ bs, const float* __restrict__ emb,
                _Float16* __restrict__ eq16, _Float16* __restrict__ ed16)
{
    int idx = blockIdx.x;            // b*272 + r
    int b = idx / ROWS_, r = idx % ROWS_;
    int t = threadIdx.x;             // 0..319
    float v = 0.f;
    if (r < 16) {
        if (r < Q_ && t < E_) v = emb[(size_t)bq[b*Q_ + r]*E_ + t];
        eq16[((size_t)b*16 + r)*EP_ + t] = (_Float16)v;
    } else {
        int d = r - 16;
        if (d < L_) {
            if (t < E_) v = emb[(size_t)bd[b*L_ + d]*E_ + t];
        } else if (d < D_) {
            if (t < E_) {
                int c = d - L_;
                float s = 0.f;
                for (int j = 0; j < CL_; ++j)
                    s += emb[(size_t)bs[(b*C_ + c)*CL_ + j]*E_ + t];
                v = s * (1.f/CL_);
            }
        }
        ed16[((size_t)b*DP_ + d)*EP_ + t] = (_Float16)v;
    }
}

// ---------------------------------------------------------------------------
// 2. uni_w [c][e] -> fp16 padded uniW16 [304 c][320 e]  (B operand, K=e contiguous)
// ---------------------------------------------------------------------------
__global__ __launch_bounds__(256)
void prep_uniW(const float* __restrict__ uni_w, _Float16* __restrict__ uniW16)
{
    int i = blockIdx.x*256 + threadIdx.x;
    if (i >= NP_*EP_) return;
    int c = i / EP_, e = i % EP_;
    float v = (e < E_ && c < E_) ? uni_w[(size_t)c*E_ + e] : 0.f;
    uniW16[i] = (_Float16)v;
}

// ---------------------------------------------------------------------------
// 3. conv weights [50][300][3][kw] -> fp16 wPrep [64][kw][3][320] (e contiguous,
//    zero-padded in k and e so the conv kernel needs no guards on A)
// ---------------------------------------------------------------------------
__global__ __launch_bounds__(256)
void prep_wconv(const float* __restrict__ w, _Float16* __restrict__ wp, int kw)
{
    int total = 64*kw*3*EP_;
    int i = blockIdx.x*256 + threadIdx.x;
    if (i >= total) return;
    int e = i % EP_; int tmp = i / EP_;
    int dq = tmp % 3; tmp /= 3;
    int dd = tmp % kw; int k = tmp / kw;
    float v = (k < 50 && e < E_) ? w[(((size_t)k*E_ + e)*3 + dq)*kw + dd] : 0.f;
    wp[i] = (_Float16)v;
}

// ---------------------------------------------------------------------------
// 4. uni transform as WMMA GEMM: rows = [eq;ed] (272/batch), N=304, K=320
//    out = relu(X @ uni_w^T + b) + 1e-9   (fp32)
//    All fragments are contiguous 16B loads (global_load_b128).
// ---------------------------------------------------------------------------
__global__ __launch_bounds__(256)
void uni_gemm_wmma(const _Float16* __restrict__ eq16, const _Float16* __restrict__ ed16,
                   const _Float16* __restrict__ uniW16, const float* __restrict__ uni_b,
                   float* __restrict__ unif32)            // [B][272][304]
{
    int b = blockIdx.x;
    int wave = threadIdx.x >> 5, lane = threadIdx.x & 31;
    int m = lane & 15, hi = lane >> 4;
    const _Float16* eqb = eq16 + (size_t)b*16*EP_;
    const _Float16* edb = ed16 + (size_t)b*DP_*EP_;
    float* outb = unif32 + (size_t)b*ROWS_*NP_;

    for (int tile = wave; tile < 17*19; tile += 8) {
        int mt = tile / 19, nt = tile % 19;
        int row = mt*16 + m;                       // 0..271 (lane's A row)
        const _Float16* src = (row < 16) ? (eqb + (size_t)row*EP_)
                                         : (edb + (size_t)(row-16)*EP_);
        int nn = nt*16 + m;                        // lane's B/D column
        const v8h* sv = (const v8h*)src;
        const v8h* wv = (const v8h*)(uniW16 + (size_t)nn*EP_);
        v8f acc = {};
        for (int c = 0; c < EP_/32; ++c) {
            int i0 = c*4 + hi;                     // A: e0 + hi*8    (8 f16)
            v16h a  = CONCAT16(sv[i0], sv[i0+2]);  //    e0+16+hi*8
            int ib = c*4 + hi;                     // B: e0 + hi*8, e0+16+hi*8
            v16h bb = CONCAT16(wv[ib], wv[ib+2]);
            acc = __builtin_amdgcn_wmma_f32_16x16x32_f16(false, a, false, bb,
                                                         (short)0, acc, false, false);
        }
        float bias = (nn < E_) ? uni_b[nn] : 0.f;
        #pragma unroll
        for (int r = 0; r < 8; ++r) {
            int row_o = mt*16 + r + hi*8;
            float v = acc[r] + bias;
            v = (v > 0.f ? v : 0.f) + 1e-9f;
            outb[(size_t)row_o*NP_ + nn] = v;
        }
    }
}

// ---------------------------------------------------------------------------
// 5. L2-normalize each uni-output row -> xn16 [B][272][320] fp16
// ---------------------------------------------------------------------------
__global__ __launch_bounds__(320)
void normalize_rows(const float* __restrict__ unif32, _Float16* __restrict__ xn16)
{
    int row = blockIdx.x, b = blockIdx.y;
    const float* src = unif32 + ((size_t)b*ROWS_ + row)*NP_;
    _Float16*   dst = xn16   + ((size_t)b*ROWS_ + row)*EP_;
    __shared__ float red[320];
    __shared__ float ssum;
    int t = threadIdx.x;
    float v = (t < E_) ? src[t] : 0.f;
    red[t] = v*v;
    __syncthreads();
    if (t < 64) red[t] += red[t + 256];
    __syncthreads();
    for (int st = 128; st > 0; st >>= 1) {
        if (t < st) red[t] += red[t + st];
        __syncthreads();
    }
    if (t == 0) ssum = red[0];
    __syncthreads();
    float sc = 1.f / fmaxf(sqrtf(ssum), 1e-10f);
    dst[t] = (_Float16)((t < E_) ? v*sc : 0.f);
}

// ---------------------------------------------------------------------------
// 6. gate = softmax_q(eq @ gate_w + gate_b) -> gate [B][16]
// ---------------------------------------------------------------------------
__global__ __launch_bounds__(320)
void gate_kernel(const _Float16* __restrict__ eq16, const float* __restrict__ gate_w,
                 const float* __restrict__ gate_b, float* __restrict__ gate)
{
    int b = blockIdx.x, t = threadIdx.x;
    __shared__ float red[320];
    __shared__ float logits[Q_];
    for (int q = 0; q < Q_; ++q) {
        float v = (t < E_) ? (float)eq16[((size_t)b*16 + q)*EP_ + t] * gate_w[t] : 0.f;
        red[t] = v;
        __syncthreads();
        if (t < 64) red[t] += red[t + 256];
        __syncthreads();
        for (int st = 128; st > 0; st >>= 1) {
            if (t < st) red[t] += red[t + st];
            __syncthreads();
        }
        if (t == 0) logits[q] = red[0] + gate_b[0];
        __syncthreads();
    }
    if (t == 0) {
        float mx = -1e30f;
        for (int q = 0; q < Q_; ++q) mx = fmaxf(mx, logits[q]);
        float s = 0.f, ex[Q_];
        for (int q = 0; q < Q_; ++q) { ex[q] = __expf(logits[q]-mx); s += ex[q]; }
        for (int q = 0; q < Q_; ++q) gate[(size_t)b*16 + q] = ex[q]/s;
    }
}

// ---------------------------------------------------------------------------
// 7. cosine sim + RBF-kernel pooling: lps [B][16][8]
// ---------------------------------------------------------------------------
__global__ __launch_bounds__(256)
void sim_lps_kernel(const _Float16* __restrict__ xn16, const float* __restrict__ mu,
                    const float* __restrict__ sigma, float* __restrict__ lps)
{
    int b = blockIdx.x / Q_, q = blockIdx.x % Q_;
    const _Float16* qn = xn16 + ((size_t)b*ROWS_ + q)*EP_;
    const _Float16* dn = xn16 + ((size_t)b*ROWS_ + 16)*EP_;
    int t = threadIdx.x;
    float s = 0.f;
    if (t < D_) {
        const v8h* qv = (const v8h*)qn;
        const v8h* dv = (const v8h*)(dn + (size_t)t*EP_);
        for (int e8 = 0; e8 < EP_/8; ++e8) {
            v8h a = qv[e8], d8 = dv[e8];
            #pragma unroll
            for (int j = 0; j < 8; ++j) s += (float)a[j]*(float)d8[j];
        }
    }
    __shared__ float red[256];
    for (int nb = 0; nb < 5; ++nb) {
        float pv = 0.f;
        if (t < D_) {
            float dlt = s - mu[nb];
            float sg  = sigma[nb];
            pv = __expf(-dlt*dlt/(sg*sg)*0.5f);
        }
        red[t] = pv;
        __syncthreads();
        for (int st = 128; st > 0; st >>= 1) {
            if (t < st) red[t] += red[t + st];
            __syncthreads();
        }
        if (t == 0) lps[((size_t)b*16 + q)*8 + nb] = logf(fmaxf(red[0], 1e-10f))*0.01f;
        __syncthreads();
    }
}

// ---------------------------------------------------------------------------
// 8. Match-tensor conv as per-(b,q) WMMA GEMM (kh folded into A on the fly):
//    A = v[k,(dd,e)] = sum_dq wPrep[k,dd,dq,e]*eq[q+dq-1,e]  (v_pk_fma_f16)
//    B = shifted ed rows (implicit im2col, b128 loads)
//    f[k,d] = relu(A @ B + bias) -> feat16 [B][150][10][256]
// ---------------------------------------------------------------------------
__global__ __launch_bounds__(256)
void conv_match_wmma(const _Float16* __restrict__ eq16, const _Float16* __restrict__ ed16,
                     const _Float16* __restrict__ wPrep, const float* __restrict__ bias,
                     _Float16* __restrict__ feat, int kw, int chanOff)
{
    const int bq = blockIdx.x;
    const int b = bq / Q_, q = bq % Q_;
    const int pad = (kw - 1) >> 1;
    const int wave = threadIdx.x >> 5, lane = threadIdx.x & 31;
    const int m = lane & 15, hi = lane >> 4;
    const int mt = wave >> 1;            // M-tile 0..3 (channels mt*16..)
    const int nh = wave & 1;             // which half of the 16 N-tiles
    const int kch = mt*16 + m;           // this lane's A-matrix row (out channel)

    const _Float16* eqb = eq16 + (size_t)b*16*EP_;
    const _Float16* edb = ed16 + (size_t)b*DP_*EP_;

    // query-row pointers for the kh=3 fold; qr<0 remapped to zero row 15
    const v8h* eqv[3];
    #pragma unroll
    for (int dq = 0; dq < 3; ++dq) {
        int qr = q + dq - 1;
        if (qr < 0) qr = 15;             // rows 10..15 of eq16 are zero
        eqv[dq] = (const v8h*)(eqb + (size_t)qr*EP_);
    }

    v8f acc[8];
    #pragma unroll
    for (int t = 0; t < 8; ++t) acc[t] = (v8f){};

    const int nChunks = kw * (EP_/32);   // each 32-chunk lies inside one dd
    for (int c = 0; c < nChunks; ++c) {
        const int dd = c / (EP_/32);
        const int e0c = (c % (EP_/32)) * 4;        // chunk base, in v8h units
        // ---- A fragment: fold kh=3 with packed-f16 FMAs, all b128 loads ----
        const v8h* wv = (const v8h*)(wPrep + (((size_t)kch*kw + dd)*3)*EP_);
        v8h s0 = {}, s1 = {};
        const int i0 = e0c + hi;                   // e0 + hi*8
        const int i1 = e0c + 2 + hi;               // e0 + 16 + hi*8
        #pragma unroll
        for (int dq = 0; dq < 3; ++dq) {
            const v8h* wrow = wv + (size_t)dq*(EP_/8);
            s0 += wrow[i0] * eqv[dq][i0];
            s1 += wrow[i1] * eqv[dq][i1];
        }
        v16h afrag = CONCAT16(s0, s1);
        // prefetch next chunk's weight stream
        {
            int cn = (c + 1 < nChunks) ? c + 1 : c;
            int ddn = cn / (EP_/32);
            int e0n = (cn % (EP_/32)) * 32;
            __builtin_prefetch(wPrep + (((size_t)kch*kw + ddn)*3)*EP_ + e0n, 0, 1);
        }
        // ---- 8 N-tiles per wave ----
        for (int t = 0; t < 8; ++t) {
            const int n0 = (nh*8 + t)*16;
            const int d  = n0 + m;                 // lane's output column
            const int ds = d + dd - pad;           // shifted doc row
            v16h bfrag;
            if (ds >= 0 && ds < D_) {
                const v8h* p = (const v8h*)(edb + (size_t)ds*EP_) + e0c + hi;
                bfrag = CONCAT16(p[0], p[2]);
            } else {
                v8h z = {};
                bfrag = CONCAT16(z, z);
            }
            acc[t] = __builtin_amdgcn_wmma_f32_16x16x32_f16(false, afrag, false, bfrag,
                                                            (short)0, acc[t], false, false);
        }
    }
    // epilogue: bias + ReLU; zero the d>=250 pad cols (pool reads them)
    for (int t = 0; t < 8; ++t) {
        const int d = (nh*8 + t)*16 + m;
        #pragma unroll
        for (int r = 0; r < 8; ++r) {
            int kl = mt*16 + r + hi*8;
            if (kl < 50) {
                float v = acc[t][r] + bias[kl];
                v = v > 0.f ? v : 0.f;
                _Float16 o = (d < D_) ? (_Float16)v : (_Float16)0.f;
                feat[((((size_t)b*150 + chanOff + kl)*Q_ + q)*DP_) + d] = o;
            }
        }
    }
}

// ---------------------------------------------------------------------------
// 9. 2x2/2 maxpool with (1,1) pad: pooled16 [B][150][6][126]
// ---------------------------------------------------------------------------
__global__ __launch_bounds__(256)
void pool_kernel(const _Float16* __restrict__ feat, _Float16* __restrict__ pooled)
{
    int i = blockIdx.x*256 + threadIdx.x;
    const int total = B_*150*6*126;
    if (i >= total) return;
    int pw = i % 126; int tmp = i / 126;
    int ph = tmp % 6; tmp /= 6;
    int ch = tmp % 150; int b = tmp / 150;
    int h0 = 2*ph - 1, w0 = 2*pw - 1;
    float mx = -1e30f;
    for (int dh = 0; dh < 2; ++dh) {
        int h = h0 + dh; if (h < 0 || h >= Q_) continue;
        for (int dw = 0; dw < 2; ++dw) {
            int wq = w0 + dw; if (wq < 0) continue;   // wq<=250 < 256, pad cols are 0
            float v = (float)feat[((((size_t)b*150 + ch)*Q_ + h)*DP_) + wq];
            mx = fmaxf(mx, v);
        }
    }
    pooled[i] = (_Float16)mx;
}

// ---------------------------------------------------------------------------
// 10. convall 150->20, 3x3, pad 1, ReLU: y [B][20][6][126] fp32
// ---------------------------------------------------------------------------
__global__ __launch_bounds__(256)
void convall_kernel(const _Float16* __restrict__ pooled, const float* __restrict__ w,
                    const float* __restrict__ bias, float* __restrict__ y)
{
    int i = blockIdx.x*256 + threadIdx.x;
    const int total = B_*20*6*126;
    if (i >= total) return;
    int wq = i % 126; int tmp = i / 126;
    int h = tmp % 6; tmp /= 6;
    int k = tmp % 20; int b = tmp / 20;
    float acc = bias[k];
    for (int c = 0; c < 150; ++c) {
        const _Float16* pb = pooled + (((size_t)b*150 + c)*6)*126;
        for (int dh = 0; dh < 3; ++dh) {
            int hh = h + dh - 1; if (hh < 0 || hh >= 6) continue;
            for (int dw = 0; dw < 3; ++dw) {
                int ww = wq + dw - 1; if (ww < 0 || ww >= 126) continue;
                acc += w[(((size_t)k*150 + c)*3 + dh)*3 + dw] * (float)pb[(size_t)hh*126 + ww];
            }
        }
    }
    y[i] = fmaxf(acc, 0.f);
}

// ---------------------------------------------------------------------------
// 11. 1x1 conv 20->30 + global spatial max: mp [B][32]
// ---------------------------------------------------------------------------
__global__ __launch_bounds__(256)
void convdim_max_kernel(const float* __restrict__ y, const float* __restrict__ cw,
                        const float* __restrict__ cb, float* __restrict__ mp)
{
    int b = blockIdx.x, t = threadIdx.x;
    __shared__ float red[256];
    const float* yb = y + (size_t)b*20*756;
    for (int s = 0; s < 30; ++s) {
        float mx = -1e30f;
        for (int pos = t; pos < 756; pos += 256) {
            float acc = cb[s];
            for (int k = 0; k < 20; ++k) acc += cw[s*20 + k]*yb[(size_t)k*756 + pos];
            mx = fmaxf(mx, acc);
        }
        red[t] = mx;
        __syncthreads();
        for (int st = 128; st > 0; st >>= 1) {
            if (t < st) red[t] = fmaxf(red[t], red[t + st]);
            __syncthreads();
        }
        if (t == 0) mp[(size_t)b*32 + s] = red[0];
        __syncthreads();
    }
}

// ---------------------------------------------------------------------------
// 12. head: out[b] = [mp | gate*lps | semantic] . out_w + out_b
// ---------------------------------------------------------------------------
__global__ __launch_bounds__(128)
void final_kernel(const float* __restrict__ mp, const float* __restrict__ gate,
                  const float* __restrict__ lps, const float* __restrict__ sem,
                  const float* __restrict__ out_w, const float* __restrict__ out_b,
                  float* __restrict__ out)
{
    int b = blockIdx.x, t = threadIdx.x;
    __shared__ float red[128];
    float v = 0.f;
    if (t < 119) {
        float x;
        if (t < 30)       x = mp[(size_t)b*32 + t];
        else if (t < 80) { int j = t - 30; int q = j/5, nb = j%5;
                           x = gate[(size_t)b*16 + q]*lps[((size_t)b*16 + q)*8 + nb]; }
        else              x = sem[(size_t)b*39 + (t - 80)];
        v = out_w[t]*x;
    }
    red[t] = v;
    __syncthreads();
    for (int st = 64; st > 0; st >>= 1) {
        if (t < st) red[t] += red[t + st];
        __syncthreads();
    }
    if (t == 0) out[b] = red[0] + out_b[0];
}

// ---------------------------------------------------------------------------
extern "C" void kernel_launch(void* const* d_in, const int* in_sizes, int n_in,
                              void* d_out, int out_size, void* d_ws, size_t ws_size,
                              hipStream_t stream)
{
    const int*   bq       = (const int*)  d_in[0];
    const int*   bd       = (const int*)  d_in[1];
    const int*   bs       = (const int*)  d_in[2];
    const float* sem      = (const float*)d_in[3];
    const float* emb      = (const float*)d_in[4];
    const float* gate_w   = (const float*)d_in[5];
    const float* gate_b   = (const float*)d_in[6];
    const float* uni_w    = (const float*)d_in[7];
    const float* uni_b    = (const float*)d_in[8];
    const float* conv1_w  = (const float*)d_in[9];
    const float* conv1_b  = (const float*)d_in[10];
    const float* conv2_w  = (const float*)d_in[11];
    const float* conv2_b  = (const float*)d_in[12];
    const float* conv3_w  = (const float*)d_in[13];
    const float* conv3_b  = (const float*)d_in[14];
    const float* convall_w= (const float*)d_in[15];
    const float* convall_b= (const float*)d_in[16];
    const float* convdim_w= (const float*)d_in[17];
    const float* convdim_b= (const float*)d_in[18];
    const float* out_w    = (const float*)d_in[19];
    const float* out_b    = (const float*)d_in[20];
    const float* mu       = (const float*)d_in[21];
    const float* sigma    = (const float*)d_in[22];
    float* out = (float*)d_out;

    // workspace carve-out (~113 MB total)
    char* ws = (char*)d_ws;
    size_t off = 0;
    auto carve = [&](size_t bytes) -> void* {
        void* p = ws + off;
        off = (off + bytes + 255) & ~(size_t)255;
        return p;
    };
    _Float16* eq16    = (_Float16*)carve((size_t)B_*16*EP_*2);
    _Float16* ed16    = (_Float16*)carve((size_t)B_*DP_*EP_*2);
    _Float16* uniW16  = (_Float16*)carve((size_t)NP_*EP_*2);
    _Float16* wp1     = (_Float16*)carve((size_t)64*3*3*EP_*2);
    _Float16* wp2     = (_Float16*)carve((size_t)64*5*3*EP_*2);
    _Float16* wp3     = (_Float16*)carve((size_t)64*7*3*EP_*2);
    float*    unif32  = (float*)   carve((size_t)B_*ROWS_*NP_*4);
    _Float16* xn16    = (_Float16*)carve((size_t)B_*ROWS_*EP_*2);
    float*    gate    = (float*)   carve((size_t)B_*16*4);
    float*    lps     = (float*)   carve((size_t)B_*16*8*4);
    _Float16* feat16  = (_Float16*)carve((size_t)B_*150*Q_*DP_*2);
    _Float16* pooled  = (_Float16*)carve((size_t)B_*150*6*126*2);
    float*    ybuf    = (float*)   carve((size_t)B_*20*6*126*4);
    float*    mp      = (float*)   carve((size_t)B_*32*4);
    (void)ws_size; (void)n_in; (void)in_sizes; (void)out_size;

    prep_embed<<<B_*ROWS_, 320, 0, stream>>>(bq, bd, bs, emb, eq16, ed16);
    prep_uniW<<<(NP_*EP_ + 255)/256, 256, 0, stream>>>(uni_w, uniW16);
    prep_wconv<<<(64*3*3*EP_ + 255)/256, 256, 0, stream>>>(conv1_w, wp1, 3);
    prep_wconv<<<(64*5*3*EP_ + 255)/256, 256, 0, stream>>>(conv2_w, wp2, 5);
    prep_wconv<<<(64*7*3*EP_ + 255)/256, 256, 0, stream>>>(conv3_w, wp3, 7);

    uni_gemm_wmma<<<B_, 256, 0, stream>>>(eq16, ed16, uniW16, uni_b, unif32);
    normalize_rows<<<dim3(ROWS_, B_), 320, 0, stream>>>(unif32, xn16);
    gate_kernel<<<B_, 320, 0, stream>>>(eq16, gate_w, gate_b, gate);
    sim_lps_kernel<<<B_*Q_, 256, 0, stream>>>(xn16, mu, sigma, lps);

    conv_match_wmma<<<B_*Q_, 256, 0, stream>>>(eq16, ed16, wp1, conv1_b, feat16, 3, 0);
    conv_match_wmma<<<B_*Q_, 256, 0, stream>>>(eq16, ed16, wp2, conv2_b, feat16, 5, 50);
    conv_match_wmma<<<B_*Q_, 256, 0, stream>>>(eq16, ed16, wp3, conv3_b, feat16, 7, 100);

    pool_kernel<<<(B_*150*6*126 + 255)/256, 256, 0, stream>>>(feat16, pooled);
    convall_kernel<<<(B_*20*6*126 + 255)/256, 256, 0, stream>>>(pooled, convall_w, convall_b, ybuf);
    convdim_max_kernel<<<B_, 256, 0, stream>>>(ybuf, convdim_w, convdim_b, mp);
    final_kernel<<<B_, 128, 0, stream>>>(mp, gate, lps, sem, out_w, out_b, out);
}
